// STDF_4638564680421
// MI455X (gfx1250) — matrix operand order repaired and checked by reading, not compiled, not run on hardware
//
#include <hip/hip_runtime.h>
#include <hip/hip_bf16.h>

// ---------------------------------------------------------------------------
// Types for gfx1250 WMMA (wave32) + TDM descriptor groups
// ---------------------------------------------------------------------------
typedef __attribute__((ext_vector_type(16))) __bf16 v16bf;
typedef __attribute__((ext_vector_type(8)))  float  v8f;
typedef __attribute__((ext_vector_type(4)))  unsigned int v4u;
typedef __attribute__((ext_vector_type(8)))  int    v8i;
typedef __attribute__((ext_vector_type(4)))  int    v4i;

__device__ __forceinline__ unsigned short f32_to_bf16_bits(float f) {
  union { float f; unsigned u; } v; v.f = f;
  unsigned u = v.u;
  u += 0x7FFFu + ((u >> 16) & 1u);      // round-to-nearest-even
  return (unsigned short)(u >> 16);
}
__device__ __forceinline__ __bf16 to_bf16(float f) {
  unsigned short s = f32_to_bf16_bits(f);
  return __builtin_bit_cast(__bf16, s);
}

#define CONV_WAVES 4
#define MTILE 64     // output channels per wave (4 x 16 WMMA subtiles)
#define NTILE 32     // pixels per wave (2 x 16 WMMA subtiles)

// ---------------------------------------------------------------------------
// Implicit-GEMM conv (stride 1, square kernel, zero pad, dilation).
// Per wave: D[64 oc x 32 pix] += A[64 oc x 32 k] * B[32 k x 32 pix], K=C*ks*ks
// => 8 WMMAs per 32-wide K chunk.
//  - A (weights): dense 2D tile (rows stride K) -> staged fp32 into LDS by the
//    Tensor Data Mover (TENSOR_LOAD_TO_LDS, 2D D#). OOB rows/cols read as zero,
//    which implements our partial-tile zero padding in the DMA engine.
//    The wave overlaps the DMA with the im2col VALU gather, then waits on
//    TENSORcnt before fragment assembly.
//  - B (im2col): gather, staged bf16 by the VALU with incremental (ci,kh,kw).
// flags: bit0 = accumulate into out, bit1 = relu, bit2 = add bias[oc]
// ---------------------------------------------------------------------------
__global__ __launch_bounds__(128) void stdf_conv_wmma(
    const float* __restrict__ x, const float* __restrict__ wgt,
    const float* __restrict__ bias, float* __restrict__ out,
    int B, int C, int H, int W, int O, int ks, int pad, int dil, int flags)
{
  const int HW  = H * W;
  const int P   = B * HW;
  const int ks2 = ks * ks;
  const int K   = C * ks2;
  const int mT  = (O + MTILE - 1) / MTILE;
  const int nT  = (P + NTILE - 1) / NTILE;
  const int wave = threadIdx.x >> 5;
  const int lane = threadIdx.x & 31;
  const int half = lane >> 4;
  const int lcol = lane & 15;
  const int tile = blockIdx.x * CONV_WAVES + wave;
  const bool active = tile < mT * nT;
  const int mBase = active ? (tile % mT) * MTILE : 0;
  const int nBase = active ? (tile / mT) * NTILE : 0;

  __shared__ float          Asf[CONV_WAVES][MTILE][32];  // fp32, filled by TDM
  __shared__ unsigned short Bs[CONV_WAVES][32][NTILE];   // bf16 bits

  // Staging pixel for this lane (one column of B): decompose once.
  const int sp = nBase + lane;
  const bool spv = sp < P;
  int sb = 0, sh = 0, sw = 0;
  if (spv) { sb = sp / HW; int hw = sp - sb * HW; sh = hw / W; sw = hw - sh * W; }
  const float* xb = x + (long)sb * C * HW;

  // Epilogue pixels for this lane (2 columns): decompose once.
  int epb[2], ephw[2];
  bool epv[2];
  #pragma unroll
  for (int n = 0; n < 2; ++n) {
    int p = nBase + n * 16 + lcol;
    epv[n] = p < P;
    int bb = epv[n] ? p / HW : 0;
    epb[n] = bb; ephw[n] = p - bb * HW;
  }

  const unsigned ldsA = (unsigned)(uintptr_t)(&Asf[wave][0][0]);

  v8f acc[8];
  #pragma unroll
  for (int i = 0; i < 8; ++i) acc[i] = (v8f){0.f,0.f,0.f,0.f,0.f,0.f,0.f,0.f};

  for (int kk = 0; kk < K; kk += 32) {
    if (active) {
      // ---- A: weight tile via Tensor Data Mover ----
      // D# group0: [1:0]count=1 | [63:32]lds_addr | [120:64]global_addr |
      //            [127:126]type=2
      unsigned long ga = (unsigned long)(uintptr_t)(wgt + (long)mBase * K + kk);
      unsigned Krem = (unsigned)(K - kk);
      unsigned Orem = (unsigned)(O - mBase);
      v4u g0;
      g0[0] = 1u;
      g0[1] = ldsA;
      g0[2] = (unsigned)(ga & 0xFFFFFFFFu);
      g0[3] = (unsigned)((ga >> 32) & 0x1FFFFFFu) | (2u << 30);
      // D# group1: data_size=4B; tensor_dim0=K-kk, tensor_dim1=O-mBase;
      //            tile_dim0=32, tile_dim1=64; tensor_dim0_stride=K
      v8i g1;
      g1[0] = (int)(2u << 16);
      g1[1] = (int)((Krem & 0xFFFFu) << 16);
      g1[2] = (int)(((Krem >> 16) & 0xFFFFu) | ((Orem & 0xFFFFu) << 16));
      g1[3] = (int)(((Orem >> 16) & 0xFFFFu) | (32u << 16));
      g1[4] = 64;                       // tile_dim1=64, tile_dim2=0
      g1[5] = (int)(unsigned)K;         // tensor_dim0_stride (data_size units)
      g1[6] = 0;
      g1[7] = 0;
      v4i gz  = {0, 0, 0, 0};
      v8i gz8 = {0, 0, 0, 0, 0, 0, 0, 0};
      // amdgpu-toolchain (clang-23) 6-arg form
      __builtin_amdgcn_tensor_load_to_lds(g0, g1, gz, gz, gz8, 0);

      // ---- B: im2col, 32 k-rows for this lane's pixel column (overlaps DMA) --
      int ci = kk / ks2;
      int rem = kk - ci * ks2;
      int kh = rem / ks, kw = rem - kh * ks;
      #pragma unroll 4
      for (int r = 0; r < 32; ++r) {
        float vb = 0.f;
        if (spv && (kk + r) < K) {
          int ih = sh + kh * dil - pad;
          int iw = sw + kw * dil - pad;
          if (ih >= 0 && ih < H && iw >= 0 && iw < W)
            vb = xb[(long)ci * HW + ih * W + iw];
        }
        Bs[wave][r][lane] = f32_to_bf16_bits(vb);
        if (++kw == ks) { kw = 0; if (++kh == ks) { kh = 0; ++ci; } }
      }
    }
    __syncthreads();
    if (active) {
      __builtin_amdgcn_s_wait_tensorcnt(0);   // A tile landed in LDS
      // B fragments: layout 32x16, lanes 0-15 hold K=0..15 (2/VGPR),
      // lanes 16-31 hold K=16..31; N = lane&15.
      v16bf bf0, bf1;
      #pragma unroll
      for (int e = 0; e < 16; ++e) {
        int kB = (half << 4) + e;
        bf0[e] = __builtin_bit_cast(__bf16, Bs[wave][kB][lcol]);
        bf1[e] = __builtin_bit_cast(__bf16, Bs[wave][kB][16 + lcol]);
      }
      #pragma unroll
      for (int s = 0; s < 4; ++s) {
        // A fragment: lane(M=lane&15); elem e -> K = half*16 + e
        v16bf af;
        #pragma unroll
        for (int e = 0; e < 16; ++e) {
          int kA = (half << 4) + e;
          af[e] = to_bf16(Asf[wave][s * 16 + lcol][kA]);
        }
        acc[s * 2 + 0] = __builtin_amdgcn_wmma_f32_16x16x32_bf16(
            false, af, false, bf0, (short)0, acc[s * 2 + 0], false, false);
        acc[s * 2 + 1] = __builtin_amdgcn_wmma_f32_16x16x32_bf16(
            false, af, false, bf1, (short)0, acc[s * 2 + 1], false, false);
      }
    }
    __syncthreads();
  }

  if (active) {
    #pragma unroll
    for (int n = 0; n < 2; ++n) {
      if (!epv[n]) continue;
      const long obase = (long)epb[n] * O;
      const int hw = ephw[n];
      #pragma unroll
      for (int s = 0; s < 4; ++s) {
        #pragma unroll
        for (int v = 0; v < 8; ++v) {
          int oc = mBase + s * 16 + v + (half << 3);  // C/D: VGPR v -> M=v (+8 hi half)
          if (oc < O) {
            long oidx = (obase + oc) * HW + hw;
            float r = acc[s * 2 + n][v];
            if (flags & 4) r += bias[oc];
            if (flags & 1) r += out[oidx];
            if (flags & 2) r = fmaxf(r, 0.f);
            out[oidx] = r;
          }
        }
      }
    }
  }
}

// ---------------------------------------------------------------------------
// BatchNorm: batch statistics over (N,H,W) per channel; biased variance.
// ---------------------------------------------------------------------------
__global__ __launch_bounds__(256) void bn_stats_kernel(
    const float* __restrict__ x, float* __restrict__ stats, int B, int C, int HW)
{
  const int c = blockIdx.x;
  const long n = (long)B * HW;
  float s = 0.f, s2 = 0.f;
  for (long i = threadIdx.x; i < n; i += blockDim.x) {
    long b = i / HW, r = i - b * HW;
    float v = x[(b * C + c) * (long)HW + r];
    s += v; s2 += v * v;
  }
  __shared__ float sh[256], sh2[256];
  sh[threadIdx.x] = s; sh2[threadIdx.x] = s2;
  __syncthreads();
  for (int st = 128; st > 0; st >>= 1) {
    if ((int)threadIdx.x < st) {
      sh[threadIdx.x]  += sh[threadIdx.x + st];
      sh2[threadIdx.x] += sh2[threadIdx.x + st];
    }
    __syncthreads();
  }
  if (threadIdx.x == 0) {
    float m   = sh[0] / (float)n;
    float var = sh2[0] / (float)n - m * m;
    stats[c]     = m;
    stats[C + c] = rsqrtf(fmaxf(var, 0.f) + 1e-5f);
  }
}

__global__ void bn_apply_kernel(const float* __restrict__ x, float* __restrict__ y,
    const float* __restrict__ stats, const float* __restrict__ g,
    const float* __restrict__ be, int B, int C, int HW, int relu)
{
  long n = (long)B * C * HW;
  long stride = (long)gridDim.x * blockDim.x;
  for (long i = (long)blockIdx.x * blockDim.x + threadIdx.x; i < n; i += stride) {
    int c = (int)((i / HW) % C);
    float v = (x[i] - stats[c]) * stats[C + c] * g[c] + be[c];
    if (relu) v = fmaxf(v, 0.f);
    y[i] = v;
  }
}

// ---------------------------------------------------------------------------
// 2x2 max pool, stride 2
// ---------------------------------------------------------------------------
__global__ void maxpool_kernel(const float* __restrict__ x, float* __restrict__ y,
                               int B, int C, int H, int W)
{
  int Ho = H >> 1, Wo = W >> 1;
  long n = (long)B * C * Ho * Wo;
  long stride = (long)gridDim.x * blockDim.x;
  for (long i = (long)blockIdx.x * blockDim.x + threadIdx.x; i < n; i += stride) {
    int wo = (int)(i % Wo); long r = i / Wo;
    int ho = (int)(r % Ho); r /= Ho;
    int c  = (int)(r % C);  int b = (int)(r / C);
    const float* xp = x + ((long)b * C + c) * H * W + (long)(ho * 2) * W + wo * 2;
    y[i] = fmaxf(fmaxf(xp[0], xp[1]), fmaxf(xp[W], xp[W + 1]));
  }
}

// ---------------------------------------------------------------------------
// 2x bilinear upsample, align_corners=True; writes into dest at channel base cb
// (fuses the concat). Separable interp == product-form bilinear.
// ---------------------------------------------------------------------------
__global__ void upsample_kernel(const float* __restrict__ x, float* __restrict__ y,
                                int B, int C, int H, int W, int Cd, int cb)
{
  int Ho = 2 * H, Wo = 2 * W;
  float shs = (Ho > 1) ? (float)(H - 1) / (float)(Ho - 1) : 0.f;
  float sws = (Wo > 1) ? (float)(W - 1) / (float)(Wo - 1) : 0.f;
  long n = (long)B * C * Ho * Wo;
  long stride = (long)gridDim.x * blockDim.x;
  for (long i = (long)blockIdx.x * blockDim.x + threadIdx.x; i < n; i += stride) {
    int wo = (int)(i % Wo); long r = i / Wo;
    int ho = (int)(r % Ho); r /= Ho;
    int c  = (int)(r % C);  int b = (int)(r / C);
    float ph = ho * shs, pw = wo * sws;
    int h0 = (int)ph, w0 = (int)pw;
    int h1 = h0 + 1 < H ? h0 + 1 : H - 1;
    int w1 = w0 + 1 < W ? w0 + 1 : W - 1;
    float fh = ph - (float)h0, fw = pw - (float)w0;
    const float* xp = x + ((long)b * C + c) * H * W;
    float v = xp[h0 * W + w0] * (1.f - fh) * (1.f - fw)
            + xp[h0 * W + w1] * (1.f - fh) * fw
            + xp[h1 * W + w0] * fh * (1.f - fw)
            + xp[h1 * W + w1] * fh * fw;
    y[(((long)b * Cd + cb + c) * Ho + ho) * (long)Wo + wo] = v;
  }
}

// Copy C channels of x into dest (Cd total channels) at channel base cb.
__global__ void copych_kernel(const float* __restrict__ x, float* __restrict__ y,
                              int B, int C, int HW, int Cd, int cb)
{
  long n = (long)B * C * HW;
  long stride = (long)gridDim.x * blockDim.x;
  for (long i = (long)blockIdx.x * blockDim.x + threadIdx.x; i < n; i += stride) {
    int hw = (int)(i % HW); long r = i / HW;
    int c  = (int)(r % C);  int b = (int)(r / C);
    y[((long)b * Cd + cb + c) * (long)HW + hw] = x[i];
  }
}

// xio = relu(xio + y)
__global__ void residual_relu_kernel(float* __restrict__ xio,
                                     const float* __restrict__ y, long n)
{
  long stride = (long)gridDim.x * blockDim.x;
  for (long i = (long)blockIdx.x * blockDim.x + threadIdx.x; i < n; i += stride)
    xio[i] = fmaxf(xio[i] + y[i], 0.f);
}

// ---------------------------------------------------------------------------
// Modulated deformable conv (DCNv2), G=7 groups, Cg=1, k=3, pad=1, O=64.
// One 64-thread block per output pixel: threads 0..62 compute the 63 modulated
// bilinear samples (shared across output channels) into LDS, then each thread
// does the 63-tap dot for its output channel. Offsets Hardtanh'd, mask sigmoid.
// ---------------------------------------------------------------------------
__global__ __launch_bounds__(64) void deform_kernel(
    const float* __restrict__ x, const float* __restrict__ om,
    const float* __restrict__ w, const float* __restrict__ bias,
    float* __restrict__ out, int B, int H, int W)
{
  const int HW = H * W;
  int pix = blockIdx.x;
  int b = pix / HW, hw = pix - b * HW;
  int h = hw / W,  wx = hw - h * W;
  __shared__ float vs[63];
  int t = threadIdx.x;
  if (t < 63) {
    int g = t / 9, k = t - g * 9;
    const float* omb = om + (long)b * 189 * HW;
    float oy = omb[(long)((g * 9 + k) * 2 + 0) * HW + hw];
    float ox = omb[(long)((g * 9 + k) * 2 + 1) * HW + hw];
    oy = fminf(fmaxf(oy, -64.f), 64.f);
    ox = fminf(fmaxf(ox, -64.f), 64.f);
    float mk = omb[(long)(126 + g * 9 + k) * HW + hw];
    mk = 1.f / (1.f + __expf(-mk));
    float py = oy + (float)h  + (float)(k / 3 - 1);
    float px = ox + (float)wx + (float)(k % 3 - 1);
    float fy0 = floorf(py), fx0 = floorf(px);
    float fy = py - fy0, fx = px - fx0;
    int y0 = (int)fy0, x0 = (int)fx0;
    const float* xp = x + ((long)b * 7 + g) * HW;
    auto samp = [&](int iy, int ix) -> float {
      return (iy >= 0 && iy < H && ix >= 0 && ix < W) ? xp[iy * W + ix] : 0.f;
    };
    float v = samp(y0, x0)         * (1.f - fy) * (1.f - fx)
            + samp(y0, x0 + 1)     * (1.f - fy) * fx
            + samp(y0 + 1, x0)     * fy * (1.f - fx)
            + samp(y0 + 1, x0 + 1) * fy * fx;
    vs[t] = v * mk;
  }
  __syncthreads();
  float acc = bias[t];
  const float* wp = w + t * 63;            // dc_w (64,7,1,9) flattened
  #pragma unroll 9
  for (int j = 0; j < 63; ++j) acc += wp[j] * vs[j];
  out[((long)b * 64 + t) * (long)HW + hw] = fmaxf(acc, 0.f);
}

// ---------------------------------------------------------------------------
// Host orchestration
// ---------------------------------------------------------------------------
extern "C" void kernel_launch(void* const* d_in, const int* in_sizes, int n_in,
                              void* d_out, int out_size, void* d_ws, size_t ws_size,
                              hipStream_t stream) {
  (void)in_sizes; (void)n_in; (void)out_size; (void)ws_size;
  auto F = [&](int i) { return (const float*)d_in[i]; };

  // jax pytree flatten order (sorted dict keys at every level):
  // 0:inputs | 1:dc_b 2:dc_w | dil0{b1,b2,b3,w1,w2,w3}=3..8 | dil1=9..14 |
  // down1..4 {b1,b2,g1,g2,w1,w2} = 15..38 | inc = 39..44 |
  // 45:om_b 46:om_w 47:outc_b 48:outc_w | up1..4 = 49..72
  const float* IN   = F(0);
  const float* dc_b = F(1);
  const float* dc_w = F(2);
  struct EH { const float *b1, *b2, *b3, *w1, *w2, *w3; };
  EH dil0 = {F(3),  F(4),  F(5),  F(6),  F(7),  F(8)};
  EH dil1 = {F(9),  F(10), F(11), F(12), F(13), F(14)};
  struct DCp { const float *b1, *b2, *g1, *g2, *w1, *w2; };
  DCp down1 = {F(15), F(16), F(17), F(18), F(19), F(20)};
  DCp down2 = {F(21), F(22), F(23), F(24), F(25), F(26)};
  DCp down3 = {F(27), F(28), F(29), F(30), F(31), F(32)};
  DCp down4 = {F(33), F(34), F(35), F(36), F(37), F(38)};
  DCp inc   = {F(39), F(40), F(41), F(42), F(43), F(44)};
  const float* om_b   = F(45);
  const float* om_w   = F(46);
  const float* outc_b = F(47);
  const float* outc_w = F(48);
  DCp up1 = {F(49), F(50), F(51), F(52), F(53), F(54)};
  DCp up2 = {F(55), F(56), F(57), F(58), F(59), F(60)};
  DCp up3 = {F(61), F(62), F(63), F(64), F(65), F(66)};
  DCp up4 = {F(67), F(68), F(69), F(70), F(71), F(72)};

  // Workspace arena (fp32 elements)
  float* WSf = (float*)d_ws;
  const size_t L_X1 = 16777216UL;   // 4x64x256x256
  const size_t L_X2 = 4194304UL;    // 4x64x128x128
  const size_t L_X3 = 2097152UL;    // 4x128x64x64
  const size_t L_X4 = 1048576UL;    // 4x256x32x32
  const size_t L_X5 = 262144UL;     // 4x256x16x16
  const size_t L_T0 = 49545216UL;   // max(up4 concat 4x128x256^2, om 4x189x256^2)
  const size_t L_T1 = 16777216UL;
  const size_t L_T2 = 16777216UL;
  float* X1 = WSf;
  float* X2 = X1 + L_X1;
  float* X3 = X2 + L_X2;
  float* X4 = X3 + L_X3;
  float* X5 = X4 + L_X4;
  float* T0 = X5 + L_X5;
  float* T1 = T0 + L_T0;
  float* T2 = T1 + L_T1;
  float* ST = T2 + L_T2;            // bn stats: mean[C], istd[C]

  const int NB = 4;
  auto gs = [](long n) { return (int)((n + 255) / 256); };

  auto conv = [&](const float* xin, const float* w, const float* bi, float* o,
                  int C, int H, int W, int O, int ks, int pad, int dil, int flags) {
    int P = NB * H * W;
    long tiles = (long)((O + MTILE - 1) / MTILE) * ((P + NTILE - 1) / NTILE);
    int blocks = (int)((tiles + CONV_WAVES - 1) / CONV_WAVES);
    stdf_conv_wmma<<<blocks, 128, 0, stream>>>(xin, w, bi, o, NB, C, H, W, O,
                                               ks, pad, dil, flags);
  };
  auto bn = [&](const float* xin, float* o, const float* g, const float* b,
                int C, int H, int W) {
    int HW = H * W;
    bn_stats_kernel<<<C, 256, 0, stream>>>(xin, ST, NB, C, HW);
    long n = (long)NB * C * HW;
    bn_apply_kernel<<<gs(n), 256, 0, stream>>>(xin, o, ST, g, b, NB, C, HW, 1);
  };
  auto dcblk = [&](const float* xin, float* o, int C, int H, int W, int M, int O,
                   const DCp& p, float* tA, float* tB) {
    conv(xin, p.w1, nullptr, tA, C, H, W, M, 3, 1, 1, 0);
    bn(tA, tB, p.g1, p.b1, M, H, W);
    conv(tB, p.w2, nullptr, tA, M, H, W, O, 3, 1, 1, 0);
    bn(tA, o, p.g2, p.b2, O, H, W);
  };
  auto pool = [&](const float* xin, float* o, int C, int H, int W) {
    long n = (long)NB * C * (H / 2) * (W / 2);
    maxpool_kernel<<<gs(n), 256, 0, stream>>>(xin, o, NB, C, H, W);
  };

  // ---- Encoder ----
  dcblk(IN, X1, 7, 256, 256, 64, 64, inc, T1, T0);
  pool(X1, T2, 64, 256, 256);
  dcblk(T2, X2, 64, 128, 128, 64, 64, down1, T1, T0);
  pool(X2, T2, 64, 128, 128);
  dcblk(T2, X3, 64, 64, 64, 128, 128, down2, T1, T0);
  pool(X3, T2, 128, 64, 64);
  dcblk(T2, X4, 128, 32, 32, 256, 256, down3, T1, T0);
  pool(X4, T2, 256, 32, 32);
  dcblk(T2, X5, 256, 16, 16, 256, 256, down4, T1, T0);

  // ---- EHDC x2: X5 = relu(X5 + sum of 3 dilated convs) ----
  const EH* ehs[2] = {&dil0, &dil1};
  for (int e = 0; e < 2; ++e) {
    const EH& p = *ehs[e];
    conv(X5, p.w1, p.b1, T0, 256, 16, 16, 256, 3, 1, 1, 4);      // bias
    conv(X5, p.w2, p.b2, T0, 256, 16, 16, 256, 3, 2, 2, 5);      // bias|accum
    conv(X5, p.w3, p.b3, T0, 256, 16, 16, 256, 3, 4, 4, 5);      // bias|accum
    long n = (long)NB * 256 * 16 * 16;
    residual_relu_kernel<<<gs(n), 256, 0, stream>>>(X5, T0, n);
  }

  // ---- Decoder: concat([skip, up2x(x)]) -> double_conv; out lands in T2 ----
  auto upstage = [&](const float* xprev, int Cx, int Hp, int Wp,
                     const float* skip, int Cs, int M, int O, const DCp& p) {
    int H = 2 * Hp, W = 2 * Wp, HWo = H * W, Cd = Cs + Cx;
    long ns = (long)NB * Cs * HWo;
    copych_kernel<<<gs(ns), 256, 0, stream>>>(skip, T0, NB, Cs, HWo, Cd, 0);
    long nu = (long)NB * Cx * HWo;
    upsample_kernel<<<gs(nu), 256, 0, stream>>>(xprev, T0, NB, Cx, Hp, Wp, Cd, Cs);
    dcblk(T0, T2, Cd, H, W, M, O, p, T1, T2);
  };
  upstage(X5, 256, 16, 16,   X4, 256, 256, 128, up1);
  upstage(T2, 128, 32, 32,   X3, 128, 128, 64,  up2);
  upstage(T2, 64, 64, 64,    X2, 64,  64,  64,  up3);
  upstage(T2, 64, 128, 128,  X1, 64,  64,  64,  up4);

  // ---- OutConv 1x1 (+bias +relu) -> T1 ----
  conv(T2, outc_w, outc_b, T1, 64, 256, 256, 64, 1, 0, 1, 6);
  // ---- Offset/mask head: 3x3, 64 -> 189 (+bias) -> T0 ----
  conv(T1, om_w, om_b, T0, 64, 256, 256, 189, 3, 1, 1, 4);
  // ---- Modulated deformable conv + relu -> d_out ----
  deform_kernel<<<NB * 256 * 256, 64, 0, stream>>>(IN, T0, dc_w, dc_b,
                                                   (float*)d_out, NB, 256, 256);
}
